// Linformer_32521492365381
// MI455X (gfx1250) — compile-verified, matrix operand order checked by reading, not run on hardware
//
#include <hip/hip_runtime.h>
#include <hip/hip_bf16.h>

typedef _Float16 half16 __attribute__((ext_vector_type(16)));
typedef _Float16 half8  __attribute__((ext_vector_type(8)));
typedef float    v8f    __attribute__((ext_vector_type(8)));

static __device__ __forceinline__ v8f wmma_f16(half16 a, half16 b, v8f c) {
  // (neg_a, A, neg_b, B, c_mod, C, reuse_a, reuse_b)
  return __builtin_amdgcn_wmma_f32_16x16x32_f16(false, a, false, b, (short)0, c,
                                                false, false);
}

// ---------------------------------------------------------------------------
// Pack a K x N weight (or its transpose) into per-lane WMMA B-fragment tiles:
// out[((nt*(K/32)+kt)*32 + lane)*16 + e], element e of lane maps to
// K = kt*32 + (e<8 ? hi*8+e : 16+hi*8+(e-8)), N = nt*16 + (lane&15), hi=lane>>4
// ---------------------------------------------------------------------------
__global__ void pack_b(const float* __restrict__ W, _Float16* __restrict__ out,
                       int Kdim, int Ncols, int trans) {
  int tid = blockIdx.x * blockDim.x + threadIdx.x;
  int lane = tid & 31;
  int kts = Kdim >> 5;
  int kt = (tid >> 5) % kts;
  int nt = (tid >> 5) / kts;
  if (nt >= (Ncols >> 4)) return;
  int hi8 = (lane >> 4) << 3;
  int n = (nt << 4) + (lane & 15);
  int kb = kt << 5;
  half16 v;
#pragma unroll
  for (int e = 0; e < 16; ++e) {
    int k = kb + (e < 8 ? hi8 + e : 16 + hi8 + (e - 8));
    float f = trans ? W[(size_t)n * Kdim + k] : W[(size_t)k * Ncols + n];
    v[e] = (_Float16)f;
  }
  *(half16*)(out + ((size_t)(nt * kts + kt) * 32 + lane) * 16) = v;
}

// ---------------------------------------------------------------------------
// Fully fused per-32-row-stripe (8 batch elems) kernel:
//  P1: stage x stripe -> LDS A-fragments (f16)
//  P2: qkv GEMM (WMMA f16), accumulate stripe's full 3072 cols -> LDS f16
//  P3: attn_sel sigmoid + Linformer attention; scatter head-merged output
//      directly into LDS in WMMA A-fragment layout (reusing region A)
//  P4: output projection GEMM (WMMA f16) + bias -> d_out (f32)
// Block: 256 threads (8 waves). dynLDS = 64KB A-frag + 192KB qkv = 256KB.
// ---------------------------------------------------------------------------
__global__ void __launch_bounds__(256)
linformer_fused(const float* __restrict__ x, const _Float16* __restrict__ Wq,
                const _Float16* __restrict__ Wp, const float* __restrict__ bias,
                const float* __restrict__ WE, const float* __restrict__ WF,
                float* __restrict__ out, float* __restrict__ attn_out) {
  extern __shared__ char smem[];
  _Float16* Afrag = (_Float16*)smem;               // [32 kt][2 mt][32 ln][16 e]
  _Float16* qkvh  = (_Float16*)(smem + 65536);     // [32 rows][3072 cols] f16

  const int tid  = threadIdx.x;
  const int lane = tid & 31;
  const int wave = tid >> 5;
  const int lm   = lane & 15;
  const int hi   = lane >> 4;
  const size_t row_base = (size_t)blockIdx.x * 32;   // global activation row

  // ---- P1: stage x stripe into A-fragment layout (f32 -> f16) ----
#pragma unroll
  for (int j = 0; j < 16; ++j) {
    int c = tid + j * 256;           // 4096 chunks of 8 halves
    int h  = c & 1;
    int ln = (c >> 1) & 31;
    int mt = (c >> 6) & 1;
    int kt = c >> 7;
    int row = mt * 16 + (ln & 15);
    int k0  = (kt << 5) + h * 16 + ((ln >> 4) << 3);
    const float* src = x + (row_base + row) * 1024 + k0;
    float4 f0 = *(const float4*)(src);
    float4 f1 = *(const float4*)(src + 4);
    half8 hv;
    hv[0] = (_Float16)f0.x; hv[1] = (_Float16)f0.y;
    hv[2] = (_Float16)f0.z; hv[3] = (_Float16)f0.w;
    hv[4] = (_Float16)f1.x; hv[5] = (_Float16)f1.y;
    hv[6] = (_Float16)f1.z; hv[7] = (_Float16)f1.w;
    *(half8*)(Afrag + (size_t)c * 8) = hv;
  }
  __syncthreads();

  // ---- P2: qkv GEMM. Wave w covers nt in [w*24, w*24+24), groups of 4. ----
  for (int g = 0; g < 6; ++g) {
    int nt0 = wave * 24 + g * 4;
    v8f acc[2][4] = {};
    for (int kt = 0; kt < 32; ++kt) {
      half16 a0 = *(const half16*)(Afrag + (size_t)((kt * 2 + 0) * 32 + lane) * 16);
      half16 a1 = *(const half16*)(Afrag + (size_t)((kt * 2 + 1) * 32 + lane) * 16);
#pragma unroll
      for (int i = 0; i < 4; ++i) {
        half16 b = *(const half16*)(Wq +
            ((size_t)((nt0 + i) * 32 + kt) * 32 + lane) * 16);
        acc[0][i] = wmma_f16(a0, b, acc[0][i]);
        acc[1][i] = wmma_f16(a1, b, acc[1][i]);
      }
    }
#pragma unroll
    for (int mt = 0; mt < 2; ++mt)
#pragma unroll
      for (int i = 0; i < 4; ++i) {
        int col = (nt0 + i) * 16 + lm;
#pragma unroll
        for (int r = 0; r < 8; ++r)
          qkvh[(size_t)(mt * 16 + hi * 8 + r) * 3072 + col] =
              (_Float16)acc[mt][i][r];
      }
  }
  __syncthreads();

  // ---- P3a: attn_sel_sig = sigmoid(scale * q[:,1,:].k[:,2,:]) ----
#pragma unroll
  for (int j = 0; j < 8; ++j) {
    int flat = tid + j * 256;        // 2048 = 8 batches * 16h * 16g
    int bl = flat >> 8;
    int h  = (flat >> 4) & 15;
    int gg = flat & 15;
    const _Float16* qp = qkvh + (size_t)(bl * 4 + 1) * 3072 + h * 64;          // q,n=1
    const _Float16* kp = qkvh + (size_t)(bl * 4 + 2) * 3072 + 1024 + gg * 64;  // k,n=2
    float s = 0.f;
    for (int d = 0; d < 64; ++d) s += (float)qp[d] * (float)kp[d];
    attn_out[(((size_t)blockIdx.x * 8 + bl) * 16 + h) * 16 + gg] =
        1.f / (1.f + __expf(-0.125f * s));
  }

  // ---- P3b: Linformer attention per (batch, head); tid<128 owns one.
  //      Scatter merged output into Afrag region (A-fragment layout). ----
  if (tid < 128) {
    int bl = tid >> 4;
    int h  = tid & 15;
    float E[4][4], F[4][4];
#pragma unroll
    for (int p = 0; p < 4; ++p)
#pragma unroll
      for (int n = 0; n < 4; ++n) {
        E[p][n] = WE[p * 4 + n];
        F[p][n] = WF[p * 4 + n];
      }
    const size_t r0 = (size_t)(bl * 4) * 3072;
    float attn[4][4] = {};
    for (int d = 0; d < 64; ++d) {
      float qd[4], kd[4];
#pragma unroll
      for (int n = 0; n < 4; ++n) {
        qd[n] = (float)qkvh[r0 + (size_t)n * 3072 + h * 64 + d];
        kd[n] = (float)qkvh[r0 + (size_t)n * 3072 + 1024 + h * 64 + d];
      }
#pragma unroll
      for (int p = 0; p < 4; ++p) {
        float kl = E[p][0]*kd[0] + E[p][1]*kd[1] + E[p][2]*kd[2] + E[p][3]*kd[3];
#pragma unroll
        for (int n = 0; n < 4; ++n) attn[n][p] += qd[n] * kl;
      }
    }
    float aw[4][4];
#pragma unroll
    for (int n = 0; n < 4; ++n) {
      float m = -1e30f;
#pragma unroll
      for (int p = 0; p < 4; ++p) { attn[n][p] *= 0.125f; m = fmaxf(m, attn[n][p]); }
      float s = 0.f;
#pragma unroll
      for (int p = 0; p < 4; ++p) { aw[n][p] = __expf(attn[n][p] - m); s += aw[n][p]; }
      float inv = 1.f / s;
#pragma unroll
      for (int p = 0; p < 4; ++p) aw[n][p] *= inv;
    }
    for (int d = 0; d < 64; ++d) {
      float vd[4];
#pragma unroll
      for (int n = 0; n < 4; ++n)
        vd[n] = (float)qkvh[r0 + (size_t)n * 3072 + 2048 + h * 64 + d];
      float vl[4];
#pragma unroll
      for (int p = 0; p < 4; ++p)
        vl[p] = F[p][0]*vd[0] + F[p][1]*vd[1] + F[p][2]*vd[2] + F[p][3]*vd[3];
      int col = h * 64 + d;
      int kt2 = col >> 5;
      int kl_ = col & 31;
      int hia = (kl_ >> 3) & 1;
      int e   = (kl_ < 16) ? (kl_ & 7) : (8 + (kl_ & 7));
#pragma unroll
      for (int n = 0; n < 4; ++n) {
        float o = aw[n][0]*vl[0] + aw[n][1]*vl[1] + aw[n][2]*vl[2] + aw[n][3]*vl[3];
        int row = bl * 4 + n;
        int mt  = row >> 4;
        int lm2 = row & 15;
        Afrag[((size_t)(kt2 * 2 + mt) * 32 + lm2 + 16 * hia) * 16 + e] = (_Float16)o;
      }
    }
  }
  __syncthreads();

  // ---- P4: output projection: out = merged @ Wp + bias. Wave covers 8 nt. --
  for (int g2 = 0; g2 < 2; ++g2) {
    int nt0 = wave * 8 + g2 * 4;
    v8f acc[2][4] = {};
    for (int kt = 0; kt < 32; ++kt) {
      half16 a0 = *(const half16*)(Afrag + (size_t)((kt * 2 + 0) * 32 + lane) * 16);
      half16 a1 = *(const half16*)(Afrag + (size_t)((kt * 2 + 1) * 32 + lane) * 16);
#pragma unroll
      for (int i = 0; i < 4; ++i) {
        half16 b = *(const half16*)(Wp +
            ((size_t)((nt0 + i) * 32 + kt) * 32 + lane) * 16);
        acc[0][i] = wmma_f16(a0, b, acc[0][i]);
        acc[1][i] = wmma_f16(a1, b, acc[1][i]);
      }
    }
#pragma unroll
    for (int i = 0; i < 4; ++i) {
      int col = nt0 * 16 + i * 16 + lm;
      float bc = bias[col];
#pragma unroll
      for (int mt = 0; mt < 2; ++mt)
#pragma unroll
        for (int r = 0; r < 8; ++r)
          out[(row_base + mt * 16 + hi * 8 + r) * 1024 + col] =
              acc[mt][i][r] + bc;
    }
  }
}

extern "C" void kernel_launch(void* const* d_in, const int* in_sizes, int n_in,
                              void* d_out, int out_size, void* d_ws, size_t ws_size,
                              hipStream_t stream) {
  (void)in_sizes; (void)n_in; (void)out_size; (void)ws_size;
  const float* x     = (const float*)d_in[0];  // [8192,4,1024]
  const float* Wqkv  = (const float*)d_in[1];  // [1024,3072]
  const float* Wproj = (const float*)d_in[2];  // [1024,1024]
  const float* bproj = (const float*)d_in[3];  // [1024]
  const float* WE    = (const float*)d_in[4];  // [4,4]
  const float* WF    = (const float*)d_in[5];  // [4,4]

  float* out      = (float*)d_out;
  float* attn_out = out + (size_t)8192 * 4 * 1024;   // tuple: attn_sel after out

  _Float16* wq = (_Float16*)d_ws;               // 1024*3072 f16 (6.3 MB)
  _Float16* wp = wq + (size_t)1024 * 3072;      // 1024*1024 f16 (2.1 MB)

  // Pack weights into WMMA B-fragment layout (L2-resident thereafter).
  pack_b<<<768, 256, 0, stream>>>(Wqkv, wq, 1024, 3072, 0);   // 192nt*32kt*32ln
  pack_b<<<256, 256, 0, stream>>>(Wproj, wp, 1024, 1024, 1);  // 64nt*32kt*32ln

  // Fully fused qkv-GEMM + attention + output projection, 32 rows per block.
  linformer_fused<<<1024, 256, 262144, stream>>>(x, wq, wp, bproj, WE, WF,
                                                 out, attn_out);
}
